// HyperGraphEncoder_22368189678086
// MI455X (gfx1250) — compile-verified, matrix operand order checked by reading, not run on hardware
//
#include <hip/hip_runtime.h>
#include <hip/hip_bf16.h>

// ---------------------------------------------------------------------------
// Problem constants (match reference)
// ---------------------------------------------------------------------------
#define BB   256
#define HH   256
#define NA   127
#define NP   23
#define NH   11
#define DKH  64            // head dim
#define MA   (BB * NA)     // 32512 atom rows
#define MP   (BB * NP)     // 5888 pharm rows
#define MH   (BB * NH)     // 2816 hyper rows
#define G3   768           // 3*H GRU gate width
#define GPB  8             // graphs per GRU block

// ---------------------------------------------------------------------------
// WMMA types & helpers (CDNA5 gfx1250, wave32)
// ---------------------------------------------------------------------------
typedef __attribute__((ext_vector_type(16))) __bf16          v16bf;
typedef __attribute__((ext_vector_type(8)))  float           v8f;
typedef __attribute__((ext_vector_type(8)))  unsigned short  u16x8;

__device__ inline unsigned short f2bf(float x) {
  // round-to-nearest-even fp32 -> bf16 bits
  unsigned int u = __builtin_bit_cast(unsigned int, x);
  unsigned int r = u + 0x7fffu + ((u >> 16) & 1u);
  return (unsigned short)(r >> 16);
}

__device__ inline v16bf mkfrag(u16x8 lo, u16x8 hi) {
  union U { struct { u16x8 lo, hi; } s; v16bf v; } u;
  u.s.lo = lo; u.s.hi = hi;
  return u.v;
}

__device__ inline v8f vzero8() {
  v8f z = {0.f, 0.f, 0.f, 0.f, 0.f, 0.f, 0.f, 0.f};
  return z;
}

// One 16x16 (M x N) tile, K = 256, A/B bf16 rows contiguous in K.
// ap/bp are the per-lane row pointers already offset by (row*stride + khalf),
// per the CDNA5 16-bit A/B fragment layout:
//   lane l: row = l&15 ; khalf = (l>=16 ? 8 : 0)
//   frag elems 0..7  = K in [kk+khalf+0  .. kk+khalf+7 ]   (16 contiguous bytes)
//   frag elems 8..15 = K in [kk+khalf+16 .. kk+khalf+23]   (16 contiguous bytes)
__device__ inline v8f bf16_mma256(const unsigned short* ap,
                                  const unsigned short* bp, v8f acc) {
#pragma unroll
  for (int kk = 0; kk < 256; kk += 32) {
    u16x8 al = *(const u16x8*)(ap + kk);
    u16x8 ah = *(const u16x8*)(ap + kk + 16);
    u16x8 bl = *(const u16x8*)(bp + kk);
    u16x8 bh = *(const u16x8*)(bp + kk + 16);
    acc = __builtin_amdgcn_wmma_f32_16x16x32_bf16(
        /*neg_a=*/false, mkfrag(al, ah),
        /*neg_b=*/false, mkfrag(bl, bh),
        /*c_mod=*/(short)0, acc, /*reuse_a=*/false, /*reuse_b=*/false);
  }
  return acc;
}

// ---------------------------------------------------------------------------
// K0: exclusive prefix sums of the three count arrays (B=256, trivial)
// ---------------------------------------------------------------------------
__global__ void starts_kernel(const int* __restrict__ ac, const int* __restrict__ pc,
                              const int* __restrict__ hc,
                              int* __restrict__ sa, int* __restrict__ sp,
                              int* __restrict__ sh) {
  if (threadIdx.x == 0) {
    int s = 0; for (int i = 0; i < BB; ++i) { sa[i] = s; s += ac[i]; }
  } else if (threadIdx.x == 1) {
    int s = 0; for (int i = 0; i < BB; ++i) { sp[i] = s; s += pc[i]; }
  } else if (threadIdx.x == 2) {
    int s = 0; for (int i = 0; i < BB; ++i) { sh[i] = s; s += hc[i]; }
  }
}

// ---------------------------------------------------------------------------
// K1: weight prep. transpose_bf: T[j][k] = bf16(W[k][j])  (W is [H,H], y=xW)
//     conv_bf: straight fp32->bf16 copy (GRU weights are already [out,in])
// ---------------------------------------------------------------------------
__global__ void transpose_bf_kernel(const float* __restrict__ W,
                                    unsigned short* __restrict__ T) {
  int idx = blockIdx.x * blockDim.x + threadIdx.x;   // j*H + k
  int j = idx >> 8, k = idx & 255;
  T[idx] = f2bf(W[(size_t)k * HH + j]);
}

__global__ void conv_bf_kernel(const float* __restrict__ W,
                               unsigned short* __restrict__ T, int n) {
  int idx = blockIdx.x * blockDim.x + threadIdx.x;
  if (idx < n) T[idx] = f2bf(W[idx]);
}

// ---------------------------------------------------------------------------
// K2: gather + zero-pad ragged messages into dense padded arrays
//     (one block = one padded row of 256 channels)
// ---------------------------------------------------------------------------
__global__ __launch_bounds__(256) void gather_kernel(
    const float* __restrict__ amsg, const float* __restrict__ pmsg,
    const float* __restrict__ hmsg,
    const int* __restrict__ ac, const int* __restrict__ pc, const int* __restrict__ hc,
    const int* __restrict__ sa, const int* __restrict__ sp, const int* __restrict__ sh,
    float* __restrict__ atom_f, unsigned short* __restrict__ atom_bf,
    float* __restrict__ pharm_f, unsigned short* __restrict__ pharm_bf,
    unsigned short* __restrict__ hyper_bf) {
  int r = blockIdx.x, tid = threadIdx.x;
  if (r < BB * NA) {
    int b = r / NA, s = r - b * NA;
    float v = (s < ac[b]) ? amsg[(size_t)(sa[b] + s) * HH + tid] : 0.f;
    atom_f[(size_t)r * HH + tid] = v;
    atom_bf[(size_t)r * HH + tid] = f2bf(v);
  } else if (r < BB * (NA + NP)) {
    int r2 = r - BB * NA;
    int b = r2 / NP, s = r2 - b * NP;
    float v = (s < pc[b]) ? pmsg[(size_t)(sp[b] + s) * HH + tid] : 0.f;
    pharm_f[(size_t)r2 * HH + tid] = v;
    pharm_bf[(size_t)r2 * HH + tid] = f2bf(v);
  } else {
    int r3 = r - BB * (NA + NP);
    int b = r3 / NH, s = r3 - b * NH;
    float v = (s < hc[b]) ? hmsg[(size_t)(sh[b] + s) * HH + tid] : 0.f;
    hyper_bf[(size_t)r3 * HH + tid] = f2bf(v);
  }
}

// ---------------------------------------------------------------------------
// K3: generic WMMA GEMM  C[M,N] = A_bf16[M,256] x Bt_bf16[N,256]^T + bias (+res)
//     one wave = one 16x16 tile of C (wave32), 8 waves / block
// ---------------------------------------------------------------------------
__global__ __launch_bounds__(256) void gemm_bf16_kernel(
    const unsigned short* __restrict__ A, const unsigned short* __restrict__ Bt,
    const float* __restrict__ bias, const float* __restrict__ res,
    float* __restrict__ C, unsigned short* __restrict__ Cbf, int M, int N) {
  int wid = blockIdx.x * (blockDim.x >> 5) + (threadIdx.x >> 5);
  int nT = N >> 4;
  int tiles = (M >> 4) * nT;
  if (wid >= tiles) return;                 // wave-uniform; EXEC stays all-ones
  int mt = (wid / nT) << 4;
  int nt = (wid - (wid / nT) * nT) << 4;

  int lane = threadIdx.x & 31;
  int r15 = lane & 15;
  int kh = (lane >> 4) << 3;                // 0 or 8
  const unsigned short* ap = A + ((size_t)(mt + r15)) * HH + kh;
  const unsigned short* bp = Bt + ((size_t)(nt + r15)) * HH + kh;
  __builtin_prefetch(bp, 0, 1);

  v8f acc = vzero8();
  acc = bf16_mma256(ap, bp, acc);

  int rbase = mt + ((lane >> 4) << 3);      // C layout: M = r + 8*(lane>=16)
  int col = nt + r15;
  float bcol = bias[col];
#pragma unroll
  for (int rr = 0; rr < 8; ++rr) {
    int row = rbase + rr;
    float v = acc[rr] + bcol;
    if (res) v += res[(size_t)row * N + col];
    C[(size_t)row * N + col] = v;
    if (Cbf) Cbf[(size_t)row * N + col] = f2bf(v);
  }
}

// ---------------------------------------------------------------------------
// K4: masked multi-head attention core (fp32 VALU; tiny score matrices)
//     output written as bf16 (input of the following out-projection GEMM)
// ---------------------------------------------------------------------------
template <int LQ, int LK>
__global__ __launch_bounds__(256) void attn_kernel(
    const float* __restrict__ Q, const float* __restrict__ Kp,
    const float* __restrict__ Vp, const int* __restrict__ qcnt,
    const int* __restrict__ kcnt, unsigned short* __restrict__ Obf) {
  __shared__ float Ks[LK][DKH];
  __shared__ float Vs[LK][DKH];
  __shared__ float Ss[LQ][LK];
  int b = blockIdx.x, tid = threadIdx.x;
  int nq = qcnt[b], nk = kcnt[b];
  for (int hd = 0; hd < 4; ++hd) {
    for (int i = tid; i < LK * DKH; i += 256) {
      int k = i / DKH, d = i - k * DKH;
      Ks[k][d] = Kp[(size_t)(b * LK + k) * HH + hd * DKH + d];
      Vs[k][d] = Vp[(size_t)(b * LK + k) * HH + hd * DKH + d];
    }
    __syncthreads();
    for (int i = tid; i < LQ * LK; i += 256) {
      int q = i / LK, k = i - q * LK;
      const float* qr = Q + (size_t)(b * LQ + q) * HH + hd * DKH;
      float s = 0.f;
#pragma unroll 8
      for (int d = 0; d < DKH; ++d) s += qr[d] * Ks[k][d];
      Ss[q][k] = (q < nq && k < nk) ? s * 0.125f : -1e9f;   // 1/sqrt(64)
    }
    __syncthreads();
    for (int q = tid; q < LQ; q += 256) {
      float m = -3.4e38f;
      for (int k = 0; k < LK; ++k) m = fmaxf(m, Ss[q][k]);
      float sum = 0.f;
      for (int k = 0; k < LK; ++k) { float e = __expf(Ss[q][k] - m); Ss[q][k] = e; sum += e; }
      float inv = 1.f / sum;
      for (int k = 0; k < LK; ++k) Ss[q][k] *= inv;
    }
    __syncthreads();
    for (int i = tid; i < LQ * DKH; i += 256) {
      int q = i / DKH, d = i - q * DKH;
      float o = 0.f;
      for (int k = 0; k < LK; ++k) o += Ss[q][k] * Vs[k][d];
      Obf[(size_t)(b * LQ + q) * HH + hd * DKH + d] = f2bf(o);
    }
    __syncthreads();
  }
}

// ---------------------------------------------------------------------------
// K5: h0 = max over sequence positions
// ---------------------------------------------------------------------------
__global__ __launch_bounds__(256) void h0max_kernel(const float* __restrict__ h,
                                                    float* __restrict__ h0) {
  int b = blockIdx.x, j = threadIdx.x;
  float m = -3.4e38f;
  for (int t = 0; t < NA; ++t)
    m = fmaxf(m, h[(size_t)(b * NA + t) * HH + j]);
  h0[(size_t)b * HH + j] = m;
}

// ---------------------------------------------------------------------------
// K6: persistent bidirectional GRU. One block = 8 graphs x one direction, all
//     127 steps. Hidden state in LDS; both per-step GEMMs use WMMA (bf16 in,
//     fp32 accumulate); masked mean accumulated in registers.
//     LDS: hS 8KB + hsb 8KB + gbuf 24KB + hnS 8KB = 48KB.
// ---------------------------------------------------------------------------
__global__ __launch_bounds__(256) void gru_kernel(
    const float* __restrict__ h0, const unsigned short* __restrict__ xbf,
    const unsigned short* __restrict__ WihF, const unsigned short* __restrict__ WhhF,
    const unsigned short* __restrict__ WihB, const unsigned short* __restrict__ WhhB,
    const float* __restrict__ bihF, const float* __restrict__ bhhF,
    const float* __restrict__ bihB, const float* __restrict__ bhhB,
    const int* __restrict__ ac, float* __restrict__ out) {
  __shared__ __align__(16) float          hS[GPB][HH];     // hidden fp32
  __shared__ __align__(16) unsigned short hsb[16][HH];     // hidden bf16 (rows 8..15 = 0 pad)
  __shared__ __align__(16) float          gbuf[GPB][G3];   // gate pre-activations
  __shared__ __align__(16) float          hnS[GPB][HH];    // hn + bhh_n

  int dir = blockIdx.x & 1;
  int gbase = (blockIdx.x >> 1) * GPB;
  const unsigned short* Wih = dir ? WihB : WihF;
  const unsigned short* Whh = dir ? WhhB : WhhF;
  const float* bih = dir ? bihB : bihF;
  const float* bhh = dir ? bhhB : bhhF;

  int tid = threadIdx.x;
  int wv = tid >> 5;
  int lane = tid & 31;
  int r15 = lane & 15;
  int kh = (lane >> 4) << 3;
  int rowbase = (lane >> 4) << 3;

  float accv[GPB];
  int cnt[GPB];
#pragma unroll
  for (int g = 0; g < GPB; ++g) {
    hS[g][tid] = h0[(size_t)(gbase + g) * HH + tid];
    accv[g] = 0.f;
    cnt[g] = ac[gbase + g];
  }
#pragma unroll
  for (int g = GPB; g < 16; ++g) hsb[g][tid] = 0;   // zero pad rows (dummy M)
  __syncthreads();

  for (int t = 0; t < NA; ++t) {
    int pos = dir ? (NA - 1 - t) : t;

    // hidden fp32 -> bf16 fragment source
#pragma unroll
    for (int g = 0; g < GPB; ++g) hsb[g][tid] = f2bf(hS[g][tid]);
    __syncthreads();

    // phase A: gi = x_t @ Wih^T  (+ biases)
    {
      int gg = gbase + r15; if (gg > BB - 1) gg = BB - 1;   // clamp dummy rows
      const unsigned short* ap = xbf + ((size_t)gg * NA + pos) * HH + kh;
      for (int nt = wv; nt < 48; nt += 8) {
        const unsigned short* bp = Wih + ((size_t)(nt * 16 + r15)) * HH + kh;
        __builtin_prefetch(bp, 0, 1);
        v8f acc = vzero8();
        acc = bf16_mma256(ap, bp, acc);
        int colg = nt * 16 + r15;
        float base = (colg < 512) ? (bih[colg] + bhh[colg]) : bih[colg];
#pragma unroll
        for (int rr = 0; rr < 8; ++rr) {
          int row = rowbase + rr;
          if (row < GPB) gbuf[row][colg] = acc[rr] + base;
        }
      }
    }
    __syncthreads();

    // phase B: gh = h @ Whh^T ; r/z gates summed into gbuf, hn kept separate
    {
      const unsigned short* ap = &hsb[r15][kh];
      for (int nt = wv; nt < 48; nt += 8) {
        const unsigned short* bp = Whh + ((size_t)(nt * 16 + r15)) * HH + kh;
        __builtin_prefetch(bp, 0, 1);
        v8f acc = vzero8();
        acc = bf16_mma256(ap, bp, acc);
        int colg = nt * 16 + r15;
#pragma unroll
        for (int rr = 0; rr < 8; ++rr) {
          int row = rowbase + rr;
          if (row < GPB) {
            if (colg < 512) gbuf[row][colg] += acc[rr];
            else            hnS[row][colg - 512] = acc[rr] + bhh[colg];
          }
        }
      }
    }
    __syncthreads();

    // elementwise gate math + state update + masked accumulation
#pragma unroll
    for (int g = 0; g < GPB; ++g) {
      float rg = 1.f / (1.f + __expf(-gbuf[g][tid]));
      float zg = 1.f / (1.f + __expf(-gbuf[g][256 + tid]));
      float ng = tanhf(gbuf[g][512 + tid] + rg * hnS[g][tid]);
      float hv = (1.f - zg) * ng + zg * hS[g][tid];
      hS[g][tid] = hv;
      if (pos < cnt[g]) accv[g] += hv;
    }
    __syncthreads();
  }

#pragma unroll
  for (int g = 0; g < GPB; ++g)
    out[(size_t)(gbase + g) * (2 * HH) + dir * HH + tid] = accv[g] / (float)cnt[g];
}

// ---------------------------------------------------------------------------
// Host orchestration
// ---------------------------------------------------------------------------
extern "C" void kernel_launch(void* const* d_in, const int* in_sizes, int n_in,
                              void* d_out, int out_size, void* d_ws, size_t ws_size,
                              hipStream_t stream) {
  (void)in_sizes; (void)n_in; (void)out_size; (void)ws_size;

  const float* amsg = (const float*)d_in[0];
  const float* pmsg = (const float*)d_in[1];
  const float* hmsg = (const float*)d_in[2];
  const int* ac = (const int*)d_in[3];
  const int* pc = (const int*)d_in[4];
  const int* hc = (const int*)d_in[5];
  // p2h: Wq 6 bq 7 Wk 8 bk 9 Wv 10 bv 11 Wo 12 bo 13 ; a2p: 14..21
  // gru: Wih_f 22 Whh_f 23 bih_f 24 bhh_f 25 Wih_b 26 Whh_b 27 bih_b 28 bhh_b 29
  float* outp = (float*)d_out;

  // ---- workspace layout (bytes), 256B aligned, with lifetime-based aliasing
  char* ws = (char*)d_ws;
  size_t cur = 0;
  auto take = [&](size_t bytes) { size_t o = cur; cur = (cur + bytes + 255) & ~(size_t)255; return o; };

  int* sa = (int*)(ws + take(BB * 4));
  int* sp = (int*)(ws + take(BB * 4));
  int* sh = (int*)(ws + take(BB * 4));
  unsigned short* WT = (unsigned short*)(ws + take((size_t)8 * HH * HH * 2));   // 8 transposed HxH
  unsigned short* GW = (unsigned short*)(ws + take((size_t)4 * G3 * HH * 2));   // 4 GRU weights
  float* atom_f   = (float*)(ws + take((size_t)MA * HH * 4));
  float* pharm_f  = (float*)(ws + take((size_t)MP * HH * 4));
  unsigned short* atom_bf  = (unsigned short*)(ws + take((size_t)MA * HH * 2));
  unsigned short* pharm_bf = (unsigned short*)(ws + take((size_t)MP * HH * 2));
  unsigned short* hyper_bf = (unsigned short*)(ws + take((size_t)MH * HH * 2));
  float* q_p  = (float*)(ws + take((size_t)MP * HH * 4));
  float* k_h  = (float*)(ws + take((size_t)MH * HH * 4));
  float* v_h  = (float*)(ws + take((size_t)MH * HH * 4));
  unsigned short* attnP_bf = (unsigned short*)(ws + take((size_t)MP * HH * 2));
  float* q_a  = (float*)(ws + take((size_t)MA * HH * 4));
  float* k_p2 = (float*)(ws + take((size_t)MP * HH * 4));
  float* v_p2 = (float*)(ws + take((size_t)MP * HH * 4));
  float* h0s  = (float*)(ws + take((size_t)BB * HH * 4));
  // aliases (strictly later lifetimes; RMW aliasing is same-thread/same-element)
  float* pharm2            = pharm_f;                      // GEMM5: C==res elementwise
  unsigned short* pharm2_bf = pharm_bf;                    // after q_p GEMM consumed it
  unsigned short* attnA_bf = atom_bf;                      // after q_a GEMM consumed it
  float* hbuf              = atom_f;                       // GEMM10: C==res elementwise
  unsigned short* h_bf     = (unsigned short*)q_a;         // after a2p attention read q_a

  const unsigned short* WTq1 = WT + 0 * HH * HH;  // p2h Wq^T
  const unsigned short* WTk1 = WT + 1 * HH * HH;
  const unsigned short* WTv1 = WT + 2 * HH * HH;
  const unsigned short* WTo1 = WT + 3 * HH * HH;
  const unsigned short* WTq2 = WT + 4 * HH * HH;  // a2p
  const unsigned short* WTk2 = WT + 5 * HH * HH;
  const unsigned short* WTv2 = WT + 6 * HH * HH;
  const unsigned short* WTo2 = WT + 7 * HH * HH;
  unsigned short* WihF = GW + 0 * (size_t)G3 * HH;
  unsigned short* WhhF = GW + 1 * (size_t)G3 * HH;
  unsigned short* WihB = GW + 2 * (size_t)G3 * HH;
  unsigned short* WhhB = GW + 3 * (size_t)G3 * HH;

  // ---- 0. prefix sums
  starts_kernel<<<1, 32, 0, stream>>>(ac, pc, hc, sa, sp, sh);

  // ---- 1. weight prep (transpose HxH to bf16 Bt layout; convert GRU weights)
  const int wIdx[8] = {6, 8, 10, 12, 14, 16, 18, 20};
  for (int i = 0; i < 8; ++i)
    transpose_bf_kernel<<<HH * HH / 256, 256, 0, stream>>>(
        (const float*)d_in[wIdx[i]], (unsigned short*)(WT + (size_t)i * HH * HH));
  conv_bf_kernel<<<G3 * HH / 256, 256, 0, stream>>>((const float*)d_in[22], WihF, G3 * HH);
  conv_bf_kernel<<<G3 * HH / 256, 256, 0, stream>>>((const float*)d_in[23], WhhF, G3 * HH);
  conv_bf_kernel<<<G3 * HH / 256, 256, 0, stream>>>((const float*)d_in[26], WihB, G3 * HH);
  conv_bf_kernel<<<G3 * HH / 256, 256, 0, stream>>>((const float*)d_in[27], WhhB, G3 * HH);

  // ---- 2. gather/pad
  gather_kernel<<<BB * (NA + NP + NH), 256, 0, stream>>>(
      amsg, pmsg, hmsg, ac, pc, hc, sa, sp, sh,
      atom_f, atom_bf, pharm_f, pharm_bf, hyper_bf);

  auto gemm = [&](const unsigned short* A, const unsigned short* Bt, const float* bias,
                  const float* res, float* C, unsigned short* Cbf, int M, int N) {
    int tiles = (M / 16) * (N / 16);
    gemm_bf16_kernel<<<(tiles + 7) / 8, 256, 0, stream>>>(A, Bt, bias, res, C, Cbf, M, N);
  };

  // ---- 3. p2h MHA: projections, attention core, out-proj(+residual)
  gemm(pharm_bf, WTq1, (const float*)d_in[7],  nullptr, q_p, nullptr, MP, HH);
  gemm(hyper_bf, WTk1, (const float*)d_in[9],  nullptr, k_h, nullptr, MH, HH);
  gemm(hyper_bf, WTv1, (const float*)d_in[11], nullptr, v_h, nullptr, MH, HH);
  attn_kernel<NP, NH><<<BB, 256, 0, stream>>>(q_p, k_h, v_h, pc, hc, attnP_bf);
  gemm(attnP_bf, WTo1, (const float*)d_in[13], pharm_f, pharm2, pharm2_bf, MP, HH);

  // ---- 4. a2p MHA
  gemm(atom_bf,   WTq2, (const float*)d_in[15], nullptr, q_a,  nullptr, MA, HH);
  gemm(pharm2_bf, WTk2, (const float*)d_in[17], nullptr, k_p2, nullptr, MP, HH);
  gemm(pharm2_bf, WTv2, (const float*)d_in[19], nullptr, v_p2, nullptr, MP, HH);
  attn_kernel<NA, NP><<<BB, 256, 0, stream>>>(q_a, k_p2, v_p2, ac, pc, attnA_bf);
  gemm(attnA_bf, WTo2, (const float*)d_in[21], atom_f, hbuf, h_bf, MA, HH);

  // ---- 5. h0 = max over positions
  h0max_kernel<<<BB, 256, 0, stream>>>(hbuf, h0s);

  // ---- 6. persistent bidirectional GRU + masked mean -> d_out [256, 512]
  gru_kernel<<<(BB / GPB) * 2, 256, 0, stream>>>(
      h0s, h_bf, WihF, WhhF, WihB, WhhB,
      (const float*)d_in[24], (const float*)d_in[25],
      (const float*)d_in[28], (const float*)d_in[29],
      ac, outp);
}
